// PSTNetUp_72851235275219
// MI455X (gfx1250) — matrix operand order repaired and checked by reading, not compiled, not run on hardware
//
#include <hip/hip_runtime.h>
#include <hip/hip_bf16.h>

// ---------------- problem constants (from reference) ----------------
constexpr int B = 4, L = 8, N = 2048, C = 32;
constexpr int MID = 64, OUT = 128;
constexpr int M = 1024;              // N / SPATIAL_STRIDE
constexpr int KS = 32;               // K_SAMPLES
constexpr int TK = 3;                // TEMPORAL_K
constexpr int KM = TK * MID;         // 192
constexpr int F = B * L;             // 32 frames
constexpr float R2 = 0.2f * 0.2f;
constexpr float BN_EPS = 1e-5f;

typedef __bf16 bf16_t;
typedef bf16_t v16bf __attribute__((ext_vector_type(16)));
typedef unsigned short v16us __attribute__((ext_vector_type(16)));
typedef float v8f  __attribute__((ext_vector_type(8)));
typedef unsigned int u32x4 __attribute__((ext_vector_type(4)));
typedef int i32x4 __attribute__((ext_vector_type(4)));
typedef int i32x8 __attribute__((ext_vector_type(8)));

__device__ __forceinline__ unsigned short f2bf(float f) {
    unsigned int u = __float_as_uint(f);
    unsigned int r = u + 0x7FFFu + ((u >> 16) & 1u);   // round-to-nearest-even
    return (unsigned short)(r >> 16);
}

// =====================================================================
// 1) Farthest point sampling: one block per frame, coords + running
//    min-distances live in LDS (34 KB of the 320 KB WGP pool).
// =====================================================================
__global__ __launch_bounds__(256) void fps_kernel(const float* __restrict__ xyzs,
                                                  float* __restrict__ anchors) {
    const int f = blockIdx.x;                    // frame = b*L + l
    const float* pts = xyzs + (size_t)f * N * 3;
    __shared__ float sx[N], sy[N], sz[N], dist[N];
    __shared__ float redV[256];
    __shared__ int   redI[256];
    const int tid = threadIdx.x;
    for (int i = tid; i < N; i += 256) {
        sx[i] = pts[i * 3 + 0];
        sy[i] = pts[i * 3 + 1];
        sz[i] = pts[i * 3 + 2];
        dist[i] = 3.402823466e38f;
    }
    __syncthreads();
    int last = 0;
    for (int m = 0; m < M; m++) {
        const float ax = sx[last], ay = sy[last], az = sz[last];
        if (tid == 0) {
            float* a = anchors + ((size_t)f * M + m) * 3;
            a[0] = ax; a[1] = ay; a[2] = az;
        }
        float bv = -1.0f; int bi = N;
        for (int i = tid; i < N; i += 256) {
            const float dx = sx[i] - ax, dy = sy[i] - ay, dz = sz[i] - az;
            float d = fminf(dist[i], dx * dx + dy * dy + dz * dz);
            dist[i] = d;
            if (d > bv) { bv = d; bi = i; }     // strided i increasing -> first max kept
        }
        redV[tid] = bv; redI[tid] = bi;
        __syncthreads();
        for (int s = 128; s > 0; s >>= 1) {
            if (tid < s) {
                const float ov = redV[tid + s]; const int oi = redI[tid + s];
                if (ov > redV[tid] || (ov == redV[tid] && oi < redI[tid])) {
                    redV[tid] = ov; redI[tid] = oi;
                }
            }
            __syncthreads();
        }
        last = redI[0];
        __syncthreads();
    }
}

// =====================================================================
// 2) Fused ball-query + grouping + point conv (one wave per anchor).
//    g_f(32x32) @ conv_f_w^T(32x64) via 2x4 bf16 WMMA 16x16x32 tiles;
//    d-conv folded per D-element; lane-pair xor-16 reduction over k.
// =====================================================================
__global__ __launch_bounds__(128) void ball_conv_kernel(
        const float* __restrict__ xyzs, const float* __restrict__ feats,
        const float* __restrict__ conv_d_w, const float* __restrict__ conv_f_w,
        const float* __restrict__ anchors, float* __restrict__ xbuf) {
    const int wave = threadIdx.x >> 5, lane = threadIdx.x & 31;
    const int o = blockIdx.y;                         // temporal offset 0..2
    const int fidx = blockIdx.z;                      // frame
    const int b = fidx / L, l = fidx % L;
    const int m = blockIdx.x * 4 + wave;              // anchor index
    const int t = l + o - 1;                          // source time (may be pad)

    __shared__ float featLDS[4][KS][C];               // 16 KB
    __shared__ float dispLDS[4][KS][3];
    __shared__ int   idxLDS[4][KS];
    __shared__ unsigned short wfLDS[MID * C];         // conv_f_w as bf16
    __shared__ float dwLDS[MID * 3];

    for (int i = threadIdx.x; i < MID * C; i += 128) wfLDS[i] = f2bf(conv_f_w[i]);
    for (int i = threadIdx.x; i < MID * 3; i += 128) dwLDS[i] = conv_d_w[i];
    __syncthreads();

    float* xrow = xbuf + ((size_t)fidx * KM + o * MID) * M + m;
    if (t < 0 || t >= L) {
        // zero-padded frame: grouped features are all zero -> f_out == 0
        if (lane < 16)
            for (int ob = 0; ob < 4; ob++) xrow[(size_t)(ob * 16 + lane) * M] = 0.0f;
        return;
    }

    const float* nb = xyzs + ((size_t)b * L + t) * N * 3;
    const float* nf = feats + ((size_t)b * L + t) * C * N;
    const float* ac = anchors + ((size_t)fidx * M + m) * 3;
    const float ax = ac[0], ay = ac[1], az = ac[2];

    // ---- ball query: first KS in-radius indices, ascending ----
    int cnt = 0;
    for (int chunk = 0; chunk < N / 32 && cnt < KS; chunk++) {
        const int i = chunk * 32 + lane;
        const float dx = nb[i * 3 + 0] - ax, dy = nb[i * 3 + 1] - ay, dz = nb[i * 3 + 2] - az;
        unsigned int msk = (unsigned int)__ballot(dx * dx + dy * dy + dz * dz < R2);
        while (msk && cnt < KS) {
            const int bit = __ffs(msk) - 1;
            if (lane == 0) idxLDS[wave][cnt] = chunk * 32 + bit;
            cnt++;
            msk &= msk - 1;
        }
    }
    __asm__ volatile("s_wait_dscnt 0" ::: "memory");
    const int first = (cnt > 0) ? idxLDS[wave][0] : 0;
    if (lane == 0)
        for (int k = cnt; k < KS; k++) idxLDS[wave][k] = first;
    __asm__ volatile("s_wait_dscnt 0" ::: "memory");

    // ---- gather: lane = sample k ----
    const int p = idxLDS[wave][lane];
    dispLDS[wave][lane][0] = nb[p * 3 + 0] - ax;
    dispLDS[wave][lane][1] = nb[p * 3 + 1] - ay;
    dispLDS[wave][lane][2] = nb[p * 3 + 2] - az;
    for (int c = 0; c < C; c++) featLDS[wave][lane][c] = nf[(size_t)c * N + p];
    __asm__ volatile("s_wait_dscnt 0" ::: "memory");

    // ---- WMMA: f_out tiles, fused with d-conv and k-reduction ----
    const int half = lane >> 4;       // C subset selector per ISA A-layout
    const int lrow = lane & 15;
    const int cbase = half * 8;       // lanes 0-15: {0-7,16-23}; 16-31: {8-15,24-31}
    float offAcc[4] = {0.f, 0.f, 0.f, 0.f};

    for (int kb = 0; kb < 2; kb++) {
        const int krow = kb * 16 + lrow;
        v16us au;
        for (int j = 0; j < 8; j++) au[j]     = f2bf(featLDS[wave][krow][cbase + j]);
        for (int j = 0; j < 8; j++) au[8 + j] = f2bf(featLDS[wave][krow][cbase + 16 + j]);
        const v16bf a = __builtin_bit_cast(v16bf, au);
        for (int ob = 0; ob < 4; ob++) {
            const int och = ob * 16 + lrow;
            v16us bu;
            for (int j = 0; j < 8; j++) bu[j]     = wfLDS[och * C + half * 8 + j];
            for (int j = 0; j < 8; j++) bu[8 + j] = wfLDS[och * C + half * 8 + 16 + j];
            const v16bf bm = __builtin_bit_cast(v16bf, bu);
            v8f acc = {};
            acc = __builtin_amdgcn_wmma_f32_16x16x32_bf16(
                    false, a, false, bm, (short)0, acc, false, false);
            for (int v = 0; v < 8; v++) {
                const int k = kb * 16 + v + half * 8;   // D-layout: M = v + 8*(lane>=16)
                const float d = dwLDS[och * 3 + 0] * dispLDS[wave][k][0]
                              + dwLDS[och * 3 + 1] * dispLDS[wave][k][1]
                              + dwLDS[och * 3 + 2] * dispLDS[wave][k][2];
                offAcc[ob] += acc[v] * d;
            }
        }
    }
    for (int ob = 0; ob < 4; ob++) offAcc[ob] += __shfl_xor(offAcc[ob], 16, 32);
    if (lane < 16)
        for (int ob = 0; ob < 4; ob++) xrow[(size_t)(ob * 16 + lane) * M] = offAcc[ob];
}

// =====================================================================
// 3) BatchNorm statistics: per (l, channel) over B*M values.
// =====================================================================
__global__ __launch_bounds__(256) void bn_stats_kernel(
        const float* __restrict__ xbuf, const float* __restrict__ gamma,
        const float* __restrict__ beta, float* __restrict__ ss) {
    const int ch = blockIdx.x, l = blockIdx.y, tid = threadIdx.x;
    __shared__ float s1[256], s2[256];
    float a1 = 0.f, a2 = 0.f;
    for (int i = tid; i < B * M; i += 256) {
        const int bb = i >> 10, mm = i & (M - 1);
        const float v = xbuf[(((size_t)bb * L + l) * KM + ch) * M + mm];
        a1 += v; a2 += v * v;
    }
    s1[tid] = a1; s2[tid] = a2;
    __syncthreads();
    for (int s = 128; s > 0; s >>= 1) {
        if (tid < s) { s1[tid] += s1[tid + s]; s2[tid] += s2[tid + s]; }
        __syncthreads();
    }
    if (tid == 0) {
        const float inv = 1.0f / (float)(B * M);
        const float mean = s1[0] * inv;
        const float var  = s2[0] * inv - mean * mean;
        const float scale = gamma[ch] * rsqrtf(var + BN_EPS);
        ss[(l * KM + ch) * 2 + 0] = scale;
        ss[(l * KM + ch) * 2 + 1] = beta[ch] - mean * scale;
    }
}

// =====================================================================
// 4) BN apply + ReLU + pack to bf16 for the temporal GEMM.
// =====================================================================
__global__ __launch_bounds__(256) void bn_pack_kernel(
        const float* __restrict__ xbuf, const float* __restrict__ ss,
        unsigned short* __restrict__ xhat) {
    const size_t i = (size_t)blockIdx.x * 256 + threadIdx.x;
    if (i >= (size_t)B * L * KM * M) return;
    const int ch = (int)((i / M) % KM);
    const int l  = (int)((i / ((size_t)M * KM)) % L);
    const float scale = ss[(l * KM + ch) * 2 + 0];
    const float shift = ss[(l * KM + ch) * 2 + 1];
    xhat[i] = f2bf(fmaxf(xbuf[i] * scale + shift, 0.0f));
}

__global__ void pack_w_kernel(const float* __restrict__ w, unsigned short* __restrict__ wbf) {
    const int i = blockIdx.x * 256 + threadIdx.x;
    if (i < OUT * KM) wbf[i] = f2bf(w[i]);
}

// =====================================================================
// 5) Temporal GEMM: per (b,l) compute W(128x192) @ xhat(192x1024).
//    8 waves/block, one 16x16 output tile per wave.  The B slab
//    (192x16 bf16, row stride M) is DMA'd into LDS by the Tensor Data
//    Mover (unconditional this round: definitive compile probe), then
//    each wave runs 6 bf16-WMMA k-steps out of LDS.
// =====================================================================
__global__ __launch_bounds__(256) void temporal_gemm_kernel(
        const unsigned short* __restrict__ wbf, const unsigned short* __restrict__ xhat,
        float* __restrict__ out) {
    const int wave = threadIdx.x >> 5, lane = threadIdx.x & 31;
    const int colT = blockIdx.x;                 // 0..63 anchor tile
    const int l = blockIdx.y, b = blockIdx.z;
    const int fidx = b * L + l;
    const int half = lane >> 4, lr = lane & 15;
    const int row = wave * 16 + lr;              // output channel (A row)
    const int mcol = colT * 16 + lr;             // anchor column
    const unsigned short* xf = xhat + (size_t)fidx * KM * M;

    // B tile staged in LDS: 192 rows x 16 cols, row-major, bf16 (6 KB)
    __shared__ unsigned short ldsB[KM * 16];

    // ---- TDM: D# for a 2D tile (tile_dim0=16 cols, tile_dim1=192 rows)
    //      of the (KM x M) bf16 tensor; DMA global -> LDS. ----
    {
        const unsigned long long gaddr =
            (unsigned long long)(uintptr_t)(xf + (size_t)colT * 16);
        const unsigned lds_off = (unsigned)(uintptr_t)&ldsB[0];
        u32x4 g0;
        g0[0] = 1u;                                  // count=1, user mode, no gather
        g0[1] = lds_off;                             // D#.lds_addr
        g0[2] = (unsigned)(gaddr & 0xFFFFFFFFu);     // global_addr[31:0]
        g0[3] = (unsigned)((gaddr >> 32) & 0x01FFFFFFu) | (2u << 30); // [56:32] | type=2
        i32x8 g1;
        g1[0] = (int)(1u << 16);                     // wg_mask=0, data_size=1 (2B)
        g1[1] = (int)(((unsigned)M & 0xFFFFu) << 16);            // tensor_dim0[15:0] @bit48
        g1[2] = (int)((((unsigned)M >> 16) & 0xFFFFu) |
                      (((unsigned)KM & 0xFFFFu) << 16));          // dim0 hi | tensor_dim1 lo
        g1[3] = (int)((((unsigned)KM >> 16) & 0xFFFFu) |
                      (16u << 16));                               // dim1 hi | tile_dim0=16
        g1[4] = (int)((unsigned)KM & 0xFFFFu);                    // tile_dim1=192, tile_dim2=0
        g1[5] = (int)M;                                           // tensor_dim0_stride lo32
        g1[6] = 0;                                                // stride hi | dim1_stride lo
        g1[7] = 0;
        i32x4 gz = {0, 0, 0, 0};
        if (wave == 0) {                             // one TDM issue per block
#if defined(__clang_major__) && __clang_major__ >= 23
            i32x8 gz8 = {0, 0, 0, 0, 0, 0, 0, 0};
            __builtin_amdgcn_tensor_load_to_lds(g0, g1, gz, gz, gz8, 0);
#else
            __builtin_amdgcn_tensor_load_to_lds(g0, g1, gz, gz, 0);
#endif
            __builtin_amdgcn_s_wait_tensorcnt(0);    // wave0's TENSORcnt -> 0
        }
    }
    __syncthreads();                                 // publish LDS tile to all waves

    __builtin_prefetch(&wbf[row * KM], 0, 3);        // global_prefetch on A slab

    v8f acc = {};
    for (int kk = 0; kk < 6; kk++) {
        const int kbase = kk * 32;
        v16us au;
        const int cb = kbase + half * 8;             // ISA A-layout C subsets
        for (int j = 0; j < 8; j++) au[j]     = wbf[row * KM + cb + j];
        for (int j = 0; j < 8; j++) au[8 + j] = wbf[row * KM + cb + 16 + j];
        const v16bf a = __builtin_bit_cast(v16bf, au);
        v16us bu;
        for (int j = 0; j < 8; j++) bu[j]     = ldsB[(kbase + half * 8 + j) * 16 + lr];
        for (int j = 0; j < 8; j++) bu[8 + j] = ldsB[(kbase + half * 8 + 16 + j) * 16 + lr];
        const v16bf bm = __builtin_bit_cast(v16bf, bu);
        acc = __builtin_amdgcn_wmma_f32_16x16x32_bf16(
                false, a, false, bm, (short)0, acc, false, false);
    }
    for (int v = 0; v < 8; v++) {
        const int r = wave * 16 + v + half * 8;      // D-layout row
        out[((size_t)fidx * OUT + r) * M + mcol] = acc[v];
    }
}

// =====================================================================
// launch
// =====================================================================
extern "C" void kernel_launch(void* const* d_in, const int* in_sizes, int n_in,
                              void* d_out, int out_size, void* d_ws, size_t ws_size,
                              hipStream_t stream) {
    (void)in_sizes; (void)n_in; (void)out_size; (void)ws_size;
    const float* xyzs      = (const float*)d_in[0];
    const float* features  = (const float*)d_in[1];
    const float* conv_d_w  = (const float*)d_in[2];
    const float* conv_f_w  = (const float*)d_in[3];
    const float* bn_gamma  = (const float*)d_in[4];
    const float* bn_beta   = (const float*)d_in[5];
    const float* temporal_w= (const float*)d_in[6];

    float* new_xyzs = (float*)d_out;                       // (B,L,M,3) = 98304 f32
    float* new_feat = (float*)d_out + (size_t)F * M * 3;   // (B,L,OUT,M)

    // workspace carving (~38 MB)
    char* ws = (char*)d_ws;
    float* xbuf = (float*)ws;                                  ws += (size_t)B * L * KM * M * 4;
    float* ss   = (float*)ws;                                  ws += (size_t)L * KM * 2 * 4;
    unsigned short* xhat = (unsigned short*)ws;                ws += (size_t)B * L * KM * M * 2;
    unsigned short* wbf  = (unsigned short*)ws;

    fps_kernel<<<F, 256, 0, stream>>>(xyzs, new_xyzs);

    ball_conv_kernel<<<dim3(M / 4, TK, F), 128, 0, stream>>>(
        xyzs, features, conv_d_w, conv_f_w, new_xyzs, xbuf);

    bn_stats_kernel<<<dim3(KM, L), 256, 0, stream>>>(xbuf, bn_gamma, bn_beta, ss);

    const size_t nelem = (size_t)B * L * KM * M;
    bn_pack_kernel<<<(unsigned)((nelem + 255) / 256), 256, 0, stream>>>(xbuf, ss, xhat);

    pack_w_kernel<<<(OUT * KM + 255) / 256, 256, 0, stream>>>(temporal_w, wbf);

    temporal_gemm_kernel<<<dim3(M / 16, L, B), 256, 0, stream>>>(wbf, xhat, new_feat);
}
